// DAM_35493609734942
// MI455X (gfx1250) — matrix operand order repaired and checked by reading, not compiled, not run on hardware
//
#include <hip/hip_runtime.h>
#include <hip/hip_bf16.h>

// DAM training loss, fused for gfx1250 (MI455X, wave32, WMMA).
// Pipeline:
//   k_prep   : sequences f32 -> f16
//   k_phimu  : softmax(B_logits) @ memory^T -> phi_mu (f16, M x H row-major)
//   k_hatphi : per n: row-max/exp of A_logits[n] into LDS (fused A-softmax),
//              WMMA GEMM seq(256x512,f16) x E^T -> hat_phi[n] (f16), /denom
//   k_score  : per (n, 32-b tile): WMMA score GEMM vs phi_mu into 128KB LDS,
//              softmax over M fused with plus-dot + BCE -> block partial sum
//   k_reduce : deterministic tree reduction -> mean

typedef __attribute__((ext_vector_type(16))) _Float16 v16h;
typedef __attribute__((ext_vector_type(8)))  _Float16 v8h;
typedef __attribute__((ext_vector_type(8)))  float    v8f;

#define WMMA_F16(a, b, c) \
  __builtin_amdgcn_wmma_f32_16x16x32_f16(false, (a), false, (b), (short)0, (c), false, false)

__device__ __forceinline__ v16h cat8(v8h lo, v8h hi) {
  return __builtin_shufflevector(lo, hi, 0,1,2,3,4,5,6,7,8,9,10,11,12,13,14,15);
}
__device__ __forceinline__ float wave_max(float v) {
  #pragma unroll
  for (int off = 16; off; off >>= 1) v = fmaxf(v, __shfl_xor(v, off, 32));
  return v;
}
__device__ __forceinline__ float wave_sum(float v) {
  #pragma unroll
  for (int off = 16; off; off >>= 1) v += __shfl_xor(v, off, 32);
  return v;
}

// ---------------------------------------------------------------- k_prep
__global__ void k_prep(const float* __restrict__ seq, _Float16* __restrict__ seq16) {
  int i = blockIdx.x * 256 + threadIdx.x;   // 256*512 = 131072 elements
  seq16[i] = (_Float16)seq[i];
}

// ---------------------------------------------------------------- k_phimu
// One block per head h. softmax(B_logits[h]) then dot against all 1024 memory rows.
__global__ void k_phimu(const float* __restrict__ Blog, const float* __restrict__ mem,
                        _Float16* __restrict__ phi16) {
  __shared__ float w[512];
  __shared__ float red[256];
  const int h = blockIdx.x, t = threadIdx.x;
  float l0 = Blog[h * 512 + t], l1 = Blog[h * 512 + 256 + t];
  red[t] = fmaxf(l0, l1);
  __syncthreads();
  for (int s = 128; s; s >>= 1) { if (t < s) red[t] = fmaxf(red[t], red[t + s]); __syncthreads(); }
  float mx = red[0];
  __syncthreads();
  float e0 = __expf(l0 - mx), e1 = __expf(l1 - mx);
  red[t] = e0 + e1;
  __syncthreads();
  for (int s = 128; s; s >>= 1) { if (t < s) red[t] += red[t + s]; __syncthreads(); }
  float inv = 1.0f / red[0];
  w[t] = e0 * inv; w[t + 256] = e1 * inv;
  __syncthreads();
  const int wv = t >> 5, lane = t & 31;
  for (int m = wv; m < 1024; m += 8) {
    float acc = 0.f;
    for (int nn = lane; nn < 512; nn += 32) acc += w[nn] * mem[(size_t)m * 512 + nn];
    acc = wave_sum(acc);
    if (lane == 0) phi16[m * 64 + h] = (_Float16)acc;
  }
}

// ---------------------------------------------------------------- k_hatphi
// grid = 511 (n = blockIdx+1), 256 threads (8 waves).
// LDS: E[64][512] f16 (64KB) + denom[64] f32. Dynamic shared.
__global__ void k_hatphi(const float* __restrict__ Alog, const _Float16* __restrict__ seq16,
                         _Float16* __restrict__ hp16) {
  extern __shared__ __align__(32) char smem[];
  _Float16* E = (_Float16*)smem;                    // 64*512 halves
  float* denom = (float*)(smem + 64 * 512 * 2);     // 64 floats
  const int n = blockIdx.x + 1;
  const int t = threadIdx.x, wv = t >> 5, lane = t & 31;
  const float* ln = Alog + (size_t)n * 64 * 512;

  // Fused masked softmax stats: max pass, then exp into LDS + denom.
  for (int h = wv; h < 64; h += 8) {
    const float* row = ln + h * 512;
    float mx = -INFINITY;
    for (int i = lane; i < n; i += 32) mx = fmaxf(mx, row[i]);
    mx = wave_max(mx);
    float s = 0.f;
    for (int i = lane; i < 512; i += 32) {
      float e = 0.f;
      if (i < n) e = __expf(row[i] - mx);
      E[h * 512 + i] = (_Float16)e;
      s += e;
    }
    s = wave_sum(s);
    if (lane == 0) denom[h] = s;
  }
  __syncthreads();

  // GEMM: (256 b x K<=512) x (K x 64 h). Wave w owns b-tiles {2w,2w+1} x 4 h-tiles.
  const int hi = lane >> 4, lm = lane & 15;
  const int kt_max = (n + 31) >> 5;                 // masked K: E is 0 beyond n
  v8f zero = {};
  v8f acc[2][4];
  #pragma unroll
  for (int p = 0; p < 2; ++p)
    #pragma unroll
    for (int q = 0; q < 4; ++q) acc[p][q] = zero;

  for (int kt = 0; kt < kt_max; ++kt) {
    const int kb = kt * 32;
    v16h a[2];
    #pragma unroll
    for (int p = 0; p < 2; ++p) {
      const int mb = 16 * (2 * wv + p) + lm;
      const _Float16* base = seq16 + (size_t)mb * 512 + kb + 16 * hi;
      a[p] = cat8(*(const v8h*)base, *(const v8h*)(base + 8));
    }
    #pragma unroll
    for (int q = 0; q < 4; ++q) {
      const int hh = 16 * q + lm;
      const v16h bv = *(const v16h*)(E + hh * 512 + kb + 16 * hi);
      acc[0][q] = WMMA_F16(a[0], bv, acc[0][q]);
      acc[1][q] = WMMA_F16(a[1], bv, acc[1][q]);
    }
  }

  // Normalize by denom[h], store hat_phi[n-1][b][h] as f16.
  const size_t outb = (size_t)(n - 1) * 256 * 64;
  #pragma unroll
  for (int q = 0; q < 4; ++q) {
    const int hh = 16 * q + lm;
    const float dinv = 1.0f / denom[hh];
    #pragma unroll
    for (int p = 0; p < 2; ++p)
      #pragma unroll
      for (int v = 0; v < 8; ++v) {
        const int b = 16 * (2 * wv + p) + v + 8 * hi;
        hp16[outb + (size_t)b * 64 + hh] = (_Float16)(acc[p][q][v] * dinv);
      }
  }
}

// ---------------------------------------------------------------- k_score
// grid = 511*8. Block = (n, 32-batch tile). 256 threads (8 waves).
// LDS: sc[32][1024] f32 (128KB) + plus[1024] f32 + part[32]. Dynamic shared.
__global__ void k_score(const _Float16* __restrict__ hp16, const _Float16* __restrict__ phi16,
                        const float* __restrict__ mem, const float* __restrict__ seq,
                        float* __restrict__ blockSums) {
  extern __shared__ __align__(32) char smem[];
  float* sc   = (float*)smem;                       // 32*1024
  float* plus = (float*)(smem + 32 * 1024 * 4);     // 1024
  float* part = plus + 1024;                        // 32
  const int idx = blockIdx.x;
  const int n_idx = idx >> 3, bt32 = idx & 7;
  const int n = n_idx + 1, bbase = bt32 * 32;
  const int t = threadIdx.x, wv = t >> 5, lane = t & 31, hi = lane >> 4, lm = lane & 15;

  for (int m = t; m < 1024; m += 256)
    plus[m] = (mem[(size_t)m * 512 + n] > 0.f) ? 1.f : 0.f;

  // A operands: hat_phi tile (16 b x 64 h), split into two K-halves.
  const int bt = wv & 1;
  const int b_row = bbase + 16 * bt + lm;
  const _Float16* ab = hp16 + ((size_t)n_idx * 256 + b_row) * 64;
  const v16h a0 = cat8(*(const v8h*)(ab +      16 * hi), *(const v8h*)(ab + 8 + 16 * hi));
  const v16h a1 = cat8(*(const v8h*)(ab + 32 + 16 * hi), *(const v8h*)(ab + 40 + 16 * hi));

  const int mt0 = wv >> 1;
  v8f zero = {};
  for (int j = 0; j < 16; ++j) {
    const int mt = mt0 + 4 * j;
    const int m = 16 * mt + lm;
    const _Float16* bb = phi16 + (size_t)m * 64;
    const v16h b0 = *(const v16h*)(bb +      16 * hi);
    const v16h b1 = *(const v16h*)(bb + 32 + 16 * hi);
    v8f acc = zero;
    acc = WMMA_F16(a0, b0, acc);
    acc = WMMA_F16(a1, b1, acc);
    #pragma unroll
    for (int v = 0; v < 8; ++v) {
      const int bl = 16 * bt + v + 8 * hi;
      sc[bl * 1024 + 16 * mt + lm] = acc[v];       // ETA == 1.0
    }
  }
  __syncthreads();

  // Softmax over M fused with plus-dot and BCE. Wave handles 4 rows.
  #pragma unroll
  for (int q = 0; q < 4; ++q) {
    const int r = 4 * wv + q;
    const float* row = sc + r * 1024;
    float mx = -INFINITY;
    for (int m = lane; m < 1024; m += 32) mx = fmaxf(mx, row[m]);
    mx = wave_max(mx);
    float den = 0.f, num = 0.f;
    for (int m = lane; m < 1024; m += 32) {
      const float e = __expf(row[m] - mx);
      den += e; num += e * plus[m];
    }
    den = wave_sum(den); num = wave_sum(num);
    if (lane == 0) {
      float p = num / den;
      p = fminf(fmaxf(p, 1e-6f), 1.f - 1e-6f);
      const float tgt = (seq[(size_t)(bbase + r) * 512 + n] > 0.f) ? 1.f : 0.f;
      part[r] = -(tgt * __logf(p) + (1.f - tgt) * log1pf(-p));
    }
  }
  __syncthreads();
  if (t == 0) {
    float s = 0.f;
    for (int r = 0; r < 32; ++r) s += part[r];     // fixed order
    blockSums[idx] = s;
  }
}

// ---------------------------------------------------------------- k_reduce
__global__ void k_reduce(const float* __restrict__ bs, float* __restrict__ out) {
  __shared__ float red[256];
  const int t = threadIdx.x;
  float s = 0.f;
  for (int i = t; i < 511 * 8; i += 256) s += bs[i];   // fixed order -> deterministic
  red[t] = s;
  __syncthreads();
  for (int k = 128; k; k >>= 1) { if (t < k) red[t] += red[t + k]; __syncthreads(); }
  if (t == 0) out[0] = red[0] * (1.0f / (256.0f * 511.0f));
}

// ---------------------------------------------------------------- launch
extern "C" void kernel_launch(void* const* d_in, const int* in_sizes, int n_in,
                              void* d_out, int out_size, void* d_ws, size_t ws_size,
                              hipStream_t stream) {
  const float* seq  = (const float*)d_in[0];   // (256, 512)
  const float* mem  = (const float*)d_in[1];   // (1024, 512)
  const float* Alog = (const float*)d_in[2];   // (512, 64, 512)
  const float* Blog = (const float*)d_in[3];   // (64, 512)
  float* out = (float*)d_out;

  char* ws = (char*)d_ws;
  _Float16* seq16 = (_Float16*)(ws + 0);               //   256 KB
  _Float16* phi16 = (_Float16*)(ws + 262144);          //   128 KB
  _Float16* hp16  = (_Float16*)(ws + 393216);          // ~16.7 MB
  float* blockSums = (float*)(ws + 17137664);          //  16.4 KB

  const size_t lds_hatphi = 64 * 512 * 2 + 64 * 4;               // ~66 KB
  const size_t lds_score  = 32 * 1024 * 4 + 1024 * 4 + 32 * 4;   // ~135 KB
  // Opt in to >64KB dynamic LDS (gfx1250 WGP has 320KB). Host-side attr set,
  // not a stream op, so it is graph-capture safe and idempotent.
  (void)hipFuncSetAttribute((const void*)k_hatphi,
                            hipFuncAttributeMaxDynamicSharedMemorySize, (int)lds_hatphi);
  (void)hipFuncSetAttribute((const void*)k_score,
                            hipFuncAttributeMaxDynamicSharedMemorySize, (int)lds_score);

  k_prep  <<<512, 256, 0, stream>>>(seq, seq16);
  k_phimu <<<64, 256, 0, stream>>>(Blog, mem, phi16);
  k_hatphi<<<511, 256, lds_hatphi, stream>>>(Alog, seq16, hp16);
  k_score <<<511 * 8, 256, lds_score, stream>>>(hp16, phi16, mem, seq, blockSums);
  k_reduce<<<1, 256, 0, stream>>>(blockSums, out);
}
